// channel_graph_interaction_25451976196294
// MI455X (gfx1250) — matrix operand order repaired and checked by reading, not compiled, not run on hardware
//
#include <hip/hip_runtime.h>

typedef __attribute__((ext_vector_type(16))) __bf16 bf16x16;
typedef __attribute__((ext_vector_type(8)))  __bf16 bf16x8;
typedef __attribute__((ext_vector_type(8)))  float  f32x8;
typedef __attribute__((ext_vector_type(4)))  float  f32x4;

// ---------------------------------------------------------------------------
// CDNA5 async global->LDS copy (ASYNCcnt path) with inline-asm fallback
// ---------------------------------------------------------------------------
#if defined(__has_builtin)
#if __has_builtin(__builtin_amdgcn_global_load_async_to_lds_b128)
#define HAVE_ASYNC_B128 1
#endif
#if __has_builtin(__builtin_amdgcn_s_wait_asynccnt)
#define HAVE_WAIT_ASYNC 1
#endif
#endif

typedef int v4i __attribute__((vector_size(16)));
typedef __attribute__((address_space(1))) v4i* as1_v4i;
typedef __attribute__((address_space(3))) v4i* as3_v4i;

__device__ __forceinline__ void async_copy16(const __bf16* g, __bf16* lds) {
#ifdef HAVE_ASYNC_B128
  __builtin_amdgcn_global_load_async_to_lds_b128((as1_v4i)g, (as3_v4i)lds, 0, 0);
#else
  unsigned lds_off = (unsigned)(size_t)lds;   // LDS aperture: addr[31:0] is the LDS address
  asm volatile("global_load_async_to_lds_b128 %0, %1, off"
               :: "v"(lds_off), "v"(g) : "memory");
#endif
}

__device__ __forceinline__ void wait_async0() {
#ifdef HAVE_WAIT_ASYNC
  __builtin_amdgcn_s_wait_asynccnt(0);
#else
  asm volatile("s_wait_asynccnt 0" ::: "memory");
#endif
}

// ---------------------------------------------------------------------------
// WMMA helpers (v_wmma_f32_16x16x32_bf16, layouts per ISA 7.12.2)
// ---------------------------------------------------------------------------
__device__ __forceinline__ f32x8 wmma_bf16(bf16x16 a, bf16x16 b, f32x8 c) {
  return __builtin_amdgcn_wmma_f32_16x16x32_bf16(
      false, a, false, b, (short)0, c, false, false);
}

// A fragment: row-major src[row][k], K contiguous. Lane lo holds M=row0+lo.
__device__ __forceinline__ bf16x16 load_a_frag(const __bf16* __restrict__ base,
                                               int ld, int row0, int k0, int lane) {
  int lo = lane & 15, hi = lane >> 4;
  const __bf16* p = base + (size_t)(row0 + lo) * ld + k0;
  bf16x8 u = *(const bf16x8*)(p + hi * 8);
  bf16x8 v = *(const bf16x8*)(p + 16 + hi * 8);
  bf16x16 r;
#pragma unroll
  for (int i = 0; i < 8; ++i) { r[i] = u[i]; r[8 + i] = v[i]; }
  return r;
}

// B fragment from Bt[n][k]: lane lo holds N=col0+lo, K[k0+8hi .. +7] and K[k0+16+8hi .. +7]
__device__ __forceinline__ bf16x16 load_b_frag(const __bf16* __restrict__ base,
                                               int ld, int col0, int k0, int lane) {
  int lo = lane & 15, hi = lane >> 4;
  const __bf16* p = base + (size_t)(col0 + lo) * ld + k0;
  bf16x8 u = *(const bf16x8*)(p + hi * 8);
  bf16x8 v = *(const bf16x8*)(p + 16 + hi * 8);
  bf16x16 r;
#pragma unroll
  for (int i = 0; i < 8; ++i) { r[i] = u[i]; r[8 + i] = v[i]; }
  return r;
}

// A fragment out of an LDS tile with padded row stride (40 bf16 = 80B,
// so the 16 rows touched by one ds_load_b128 cover 64 distinct banks)
#define APAD 40
__device__ __forceinline__ bf16x16 lds_a_frag(const __bf16* tile, int msub, int lane) {
  int lo = lane & 15, hi = lane >> 4;
  const __bf16* p = tile + (msub + lo) * APAD;
  bf16x8 u = *(const bf16x8*)(p + hi * 8);
  bf16x8 v = *(const bf16x8*)(p + 16 + hi * 8);
  bf16x16 r;
#pragma unroll
  for (int i = 0; i < 8; ++i) { r[i] = u[i]; r[8 + i] = v[i]; }
  return r;
}

// ---------------------------------------------------------------------------
// K0a: Wt[m][k] = bf16(W[k][m])   (LDS-tiled transpose + convert)
// ---------------------------------------------------------------------------
__global__ void k_transpose_w(const float* __restrict__ W, __bf16* __restrict__ Wt) {
  __shared__ float tile[32][33];
  int m0 = blockIdx.x * 32, k0 = blockIdx.y * 32;
  int tx = threadIdx.x, ty = threadIdx.y;
  tile[ty][tx] = W[(size_t)(k0 + ty) * 4096 + (m0 + tx)];
  __syncthreads();
  Wt[(size_t)(m0 + ty) * 4096 + (k0 + tx)] = (__bf16)tile[tx][ty];
}

// K0b: Wup[(o*16+i*4+j)][c] = bf16(up_w[c][o][i][j])
__global__ void k_prep_upw(const float* __restrict__ up_w, __bf16* __restrict__ Wup) {
  int g = blockIdx.x * blockDim.x + threadIdx.x;   // 65536
  int m = g >> 6, c = g & 63;
  int o = m >> 4, ij = m & 15, i = ij >> 2, j = ij & 3;
  Wup[g] = (__bf16)up_w[(((c * 64 + o) * 4 + i) * 4) + j];
}

// K1: 4x4 maxpool: x[8,64,256,256] -> xp[8,64,64,64]
__global__ void k_maxpool(const float* __restrict__ x, float* __restrict__ xp) {
  int g = blockIdx.x * blockDim.x + threadIdx.x;   // 2,097,152
  int wo = g & 63, ho = (g >> 6) & 63, bc = g >> 12;
  const float* base = x + ((size_t)bc * 256 + ho * 4) * 256 + wo * 4;
  float m = -INFINITY;
#pragma unroll
  for (int r = 0; r < 4; ++r) {
    f32x4 v = *(const f32x4*)(base + r * 256);
    m = fmaxf(m, fmaxf(fmaxf(v[0], v[1]), fmaxf(v[2], v[3])));
  }
  xp[g] = m;
}

// K2: 3x3 conv (pad 1) + bias + relu on xp -> pih[8,64,4096]
__global__ void k_conv3x3(const float* __restrict__ xp, const float* __restrict__ pw,
                          const float* __restrict__ pb, float* __restrict__ pih) {
  int b = blockIdx.y;
  int p = blockIdx.x * 256 + threadIdx.x;
  int h = p >> 6, w = p & 63;
  float acc[64];
#pragma unroll
  for (int oc = 0; oc < 64; ++oc) acc[oc] = pb[oc];
  for (int ic = 0; ic < 64; ++ic) {
    const float* xc = xp + (size_t)(b * 64 + ic) * 4096;
    float v[9];
#pragma unroll
    for (int ky = 0; ky < 3; ++ky)
#pragma unroll
      for (int kx = 0; kx < 3; ++kx) {
        int hh = h + ky - 1, ww = w + kx - 1;
        bool ok = (hh >= 0) & (hh < 64) & (ww >= 0) & (ww < 64);
        v[ky * 3 + kx] = ok ? xc[hh * 64 + ww] : 0.0f;
      }
    const float* wic = pw + ic * 9;
#pragma unroll
    for (int oc = 0; oc < 64; ++oc) {
      const float* wo = wic + oc * 576;
      float s = acc[oc];
#pragma unroll
      for (int t = 0; t < 9; ++t) s = fmaf(wo[t], v[t], s);
      acc[oc] = s;
    }
  }
#pragma unroll
  for (int oc = 0; oc < 64; ++oc)
    pih[(size_t)(b * 64 + oc) * 4096 + p] = fmaxf(acc[oc], 0.0f);
}

// K3a: invn[b*64+c] = 1/max(||pih row||, 1e-12)
__global__ void k_norm(const float* __restrict__ pih, float* __restrict__ invn) {
  __shared__ float red[256];
  const float* row = pih + (size_t)blockIdx.x * 4096;
  float s = 0.f;
  for (int i = threadIdx.x; i < 4096; i += 256) { float v = row[i]; s = fmaf(v, v, s); }
  red[threadIdx.x] = s; __syncthreads();
  for (int k = 128; k > 0; k >>= 1) {
    if (threadIdx.x < k) red[threadIdx.x] += red[threadIdx.x + k];
    __syncthreads();
  }
  if (threadIdx.x == 0) invn[blockIdx.x] = 1.0f / fmaxf(sqrtf(red[0]), 1e-12f);
}

// K3b: nx = bf16(pih * invn)
__global__ void k_make_nx(const float* __restrict__ pih, const float* __restrict__ invn,
                          __bf16* __restrict__ nx) {
  int g = blockIdx.x * blockDim.x + threadIdx.x;
  nx[g] = (__bf16)(pih[g] * invn[g >> 12]);
}

// K4: A[b] = nx[b] * nx[b]^T   (64x64, K=4096), one wave per 16x16 tile
__global__ void k_affinity(const __bf16* __restrict__ nx, float* __restrict__ A) {
  int b = blockIdx.z;
  int m0 = blockIdx.x * 16, n0 = blockIdx.y * 16;
  int lane = threadIdx.x & 31, lo = lane & 15, hi = lane >> 4;
  const __bf16* base = nx + (size_t)b * 64 * 4096;
  f32x8 acc = {0, 0, 0, 0, 0, 0, 0, 0};
  for (int k0 = 0; k0 < 4096; k0 += 32) {
    bf16x16 a = load_a_frag(base, 4096, m0, k0, lane);
    bf16x16 bb = load_b_frag(base, 4096, n0, k0, lane);
    acc = wmma_bf16(a, bb, acc);
  }
#pragma unroll
  for (int v = 0; v < 8; ++v)
    A[(size_t)b * 4096 + (m0 + v + 8 * hi) * 64 + (n0 + lo)] = acc[v];
}

// K5: L = I - d A d with d = 1/sqrt(rowsum(A)+1e-8) (0 if non-finite)
__global__ void k_laplacian(const float* __restrict__ A, float* __restrict__ L) {
  __shared__ float dsh[64];
  int b = blockIdx.x, c = threadIdx.x;
  const float* Ab = A + (size_t)b * 4096;
  float s = 0.f;
  for (int e = 0; e < 64; ++e) s += Ab[c * 64 + e];
  float dv = rsqrtf(s + 1e-8f);
  if (!__builtin_isfinite(dv)) dv = 0.f;
  dsh[c] = dv; __syncthreads();
  float dc = dsh[c];
  for (int e = 0; e < 64; ++e)
    L[(size_t)b * 4096 + c * 64 + e] =
        ((c == e) ? 1.0f : 0.0f) - dc * Ab[c * 64 + e] * dsh[e];
}

// K5b: T1[(b*64+c)][n] = bf16( sum_d L[b][c][d] * xp[b][d][n] )   (K=64, VALU)
__global__ void k_t1(const float* __restrict__ L, const float* __restrict__ xp,
                     __bf16* __restrict__ T1) {
  int bc = blockIdx.y, b = bc >> 6;
  int n = blockIdx.x * 256 + threadIdx.x;
  const float* Lr = L + (size_t)bc * 64;
  const float* xb = xp + (size_t)b * 64 * 4096 + n;
  float s = 0.f;
#pragma unroll 8
  for (int d = 0; d < 64; ++d) s = fmaf(Lr[d], xb[(size_t)d * 4096], s);
  T1[(size_t)bc * 4096 + n] = (__bf16)s;
}

// ---------------------------------------------------------------------------
// K6: T2 = T1[512x4096] * W[4096x4096]  (bf16 WMMA, f32 acc)
// Block = 4 waves = 64m x 128n. A tile (64x32) is staged in LDS with
// double-buffered async global->LDS copies (ASYNCcnt), shared by all 4 waves.
// Each wave: 4m x 2n sub-tiles = 8 WMMA per K-step; B direct from global.
// Output written transposed + bf16 as T2t[(b*4096+hw)][c] (single 16B store
// per lane per tile), ready as the next GEMM's B operand.
// ---------------------------------------------------------------------------
__device__ __forceinline__ void stage_a(const __bf16* __restrict__ T1, int m_block,
                                        int k0, int wave, int lane, __bf16* buf) {
#pragma unroll
  for (int q = 0; q < 2; ++q) {
    int chunk = wave * 64 + q * 32 + lane;   // 0..255 -> (row, 16B chunk)
    int row = chunk >> 2, c4 = chunk & 3;
    const __bf16* g = T1 + (size_t)(m_block + row) * 4096 + k0 + c4 * 8;
    async_copy16(g, buf + row * APAD + c4 * 8);
  }
}

__global__ void k_gemm(const __bf16* __restrict__ T1, const __bf16* __restrict__ Wt,
                       __bf16* __restrict__ T2t) {
  __shared__ __bf16 Atile[2][64 * APAD];
  int wave = threadIdx.x >> 5, lane = threadIdx.x & 31;
  int lo = lane & 15, hi = lane >> 4;
  int m_block = blockIdx.y * 64;
  int n_wave = blockIdx.x * 128 + wave * 32;

  f32x8 z = {0, 0, 0, 0, 0, 0, 0, 0};
  f32x8 acc[4][2];
#pragma unroll
  for (int mi = 0; mi < 4; ++mi) { acc[mi][0] = z; acc[mi][1] = z; }

  stage_a(T1, m_block, 0, wave, lane, &Atile[0][0]);
  for (int it = 0; it < 128; ++it) {
    int k0 = it * 32;
    int buf = it & 1;
    wait_async0();                 // this wave's stage of `buf` complete
    __syncthreads();               // all waves staged `buf`; all done reading buf^1
    if (it + 1 < 128) stage_a(T1, m_block, k0 + 32, wave, lane, &Atile[buf ^ 1][0]);
    bf16x16 b0 = load_b_frag(Wt, 4096, n_wave, k0, lane);
    bf16x16 b1 = load_b_frag(Wt, 4096, n_wave + 16, k0, lane);
#pragma unroll
    for (int mi = 0; mi < 4; ++mi) {
      bf16x16 a = lds_a_frag(&Atile[buf][0], mi * 16, lane);
      acc[mi][0] = wmma_bf16(a, b0, acc[mi][0]);
      acc[mi][1] = wmma_bf16(a, b1, acc[mi][1]);
    }
  }
#pragma unroll
  for (int mi = 0; mi < 4; ++mi) {
    int mbase = m_block + mi * 16 + 8 * hi;     // 8 consecutive rows this lane owns
    int b = mbase >> 6, cbase = mbase & 63;
#pragma unroll
    for (int ni = 0; ni < 2; ++ni) {
      int n = n_wave + ni * 16 + lo;
      bf16x8 o;
#pragma unroll
      for (int v = 0; v < 8; ++v) o[v] = (__bf16)acc[mi][ni][v];
      *(bf16x8*)(T2t + (size_t)(b * 4096 + n) * 64 + cbase) = o;
    }
  }
}

// K7: ConvTranspose(4,4) as GEMM Out[(o,i,j)][(b,hw)] = Wup[1024x64]*T2t^T,
// fused + bias + identity + relu -> d_out. K=64 -> 2 WMMAs per tile.
__global__ void k_upconv_fuse(const __bf16* __restrict__ Wup, const __bf16* __restrict__ T2t,
                              const float* __restrict__ upb, const float* __restrict__ x,
                              float* __restrict__ out) {
  int wave = threadIdx.x >> 5, lane = threadIdx.x & 31;
  int lo = lane & 15, hi = lane >> 4;
  int o = blockIdx.y;
  int ntile0 = blockIdx.x * 16 + wave * 4;
  bf16x16 a0 = load_a_frag(Wup, 64, o * 16, 0, lane);
  bf16x16 a1 = load_a_frag(Wup, 64, o * 16, 32, lane);
  float bias = upb[o];
#pragma unroll
  for (int ni = 0; ni < 4; ++ni) {
    int n0 = (ntile0 + ni) * 16;
    bf16x16 b0 = load_b_frag(T2t, 64, n0, 0, lane);
    bf16x16 b1 = load_b_frag(T2t, 64, n0, 32, lane);
    f32x8 acc = {0, 0, 0, 0, 0, 0, 0, 0};
    acc = wmma_bf16(a0, b0, acc);
    acc = wmma_bf16(a1, b1, acc);
    int n = n0 + lo;
    int b = n >> 12, hw = n & 4095, h = hw >> 6, w = hw & 63;
#pragma unroll
    for (int r = 0; r < 2; ++r) {
      int idx = 8 * hi + 4 * r;                 // = i*4 + j
      int i = idx >> 2;
      size_t oidx = ((size_t)(b * 64 + o) * 256 + (4 * h + i)) * 256 + 4 * w;
      f32x4 id = *(const f32x4*)(x + oidx);
      f32x4 res;
#pragma unroll
      for (int t = 0; t < 4; ++t) res[t] = fmaxf(acc[4 * r + t] + bias + id[t], 0.0f);
      *(f32x4*)(out + oidx) = res;
    }
  }
}

// ---------------------------------------------------------------------------
// Workspace layout (bytes, 256-aligned)
// ---------------------------------------------------------------------------
static const size_t OFF_WT   = 0;          // 33554432
static const size_t OFF_XP   = 33554432;   //  8388608
static const size_t OFF_PIH  = 41943040;   //  8388608
static const size_t OFF_NX   = 50331648;   //  4194304
static const size_t OFF_INVN = 54525952;   //     2048
static const size_t OFF_A    = 54528000;   //   131072
static const size_t OFF_L    = 54659072;   //   131072
static const size_t OFF_T1   = 54790144;   //  4194304
static const size_t OFF_T2T  = 58984448;   //  4194304
static const size_t OFF_WUP  = 63178752;   //   131072
static const size_t WS_NEED  = 63309824;

extern "C" void kernel_launch(void* const* d_in, const int* in_sizes, int n_in,
                              void* d_out, int out_size, void* d_ws, size_t ws_size,
                              hipStream_t stream) {
  if (ws_size < WS_NEED) return;
  const float* x     = (const float*)d_in[0];
  const float* W     = (const float*)d_in[1];
  const float* pih_w = (const float*)d_in[2];
  const float* pih_b = (const float*)d_in[3];
  const float* up_w  = (const float*)d_in[4];
  const float* up_b  = (const float*)d_in[5];
  float* out = (float*)d_out;

  char* ws = (char*)d_ws;
  __bf16* Wt   = (__bf16*)(ws + OFF_WT);
  float*  xp   = (float*) (ws + OFF_XP);
  float*  pih  = (float*) (ws + OFF_PIH);
  __bf16* nx   = (__bf16*)(ws + OFF_NX);
  float*  invn = (float*) (ws + OFF_INVN);
  float*  Amat = (float*) (ws + OFF_A);
  float*  Lap  = (float*) (ws + OFF_L);
  __bf16* T1   = (__bf16*)(ws + OFF_T1);
  __bf16* T2t  = (__bf16*)(ws + OFF_T2T);
  __bf16* Wup  = (__bf16*)(ws + OFF_WUP);

  k_transpose_w<<<dim3(128, 128), dim3(32, 32), 0, stream>>>(W, Wt);
  k_prep_upw   <<<256, 256, 0, stream>>>(up_w, Wup);
  k_maxpool    <<<8192, 256, 0, stream>>>(x, xp);
  k_conv3x3    <<<dim3(16, 8), 256, 0, stream>>>(xp, pih_w, pih_b, pih);
  k_norm       <<<512, 256, 0, stream>>>(pih, invn);
  k_make_nx    <<<8192, 256, 0, stream>>>(pih, invn, nx);
  k_affinity   <<<dim3(4, 4, 8), 32, 0, stream>>>(nx, Amat);
  k_laplacian  <<<8, 64, 0, stream>>>(Amat, Lap);
  k_t1         <<<dim3(16, 512), 256, 0, stream>>>(Lap, xp, T1);
  k_gemm       <<<dim3(32, 8), 128, 0, stream>>>(T1, Wt, T2t);
  k_upconv_fuse<<<dim3(128, 64), 128, 0, stream>>>(Wup, T2t, up_b, x, out);
}
